// PointNetSetAbstraction_43782896615739
// MI455X (gfx1250) — compile-verified, matrix operand order checked by reading, not run on hardware
//
#include <hip/hip_runtime.h>

typedef __attribute__((ext_vector_type(16))) _Float16 v16h;
typedef __attribute__((ext_vector_type(8)))  _Float16 v8h;
typedef __attribute__((ext_vector_type(8)))  float    v8f;

#define NB     16
#define NPTS   4096
#define NPOINT 512
#define NSAMP  32
#define RAD2   0.04f
#define MROWS  (NB * NPOINT * NSAMP)   // 262144 GEMM rows

// ---------------------------------------------------------------------------
// Kernel 1: farthest point sampling. One block per batch; whole batch xyz in
// LDS (48KB of the 320KB WGP LDS). 512 sequential argmax iterations with
// wave32 shuffle butterfly + tiny LDS cross-wave reduction.
// Also writes new_xyz directly into d_out (first 16*512*3 floats).
// ---------------------------------------------------------------------------
__global__ void __launch_bounds__(512)
fps_kernel(const float* __restrict__ xyz, int* __restrict__ fpsIdx,
           float* __restrict__ outXyz)
{
    __shared__ float lx[NPTS], ly[NPTS], lz[NPTS];
    __shared__ float rv[16];
    __shared__ int   ri[16];
    __shared__ int   sFar;
    __shared__ int   sList[NPOINT];

    const int b = blockIdx.x, t = threadIdx.x;
    const float* base = xyz + (unsigned)b * NPTS * 3;
    for (int n = t; n < NPTS; n += 512) {
        lx[n] = base[3*n]; ly[n] = base[3*n+1]; lz[n] = base[3*n+2];
    }
    if (t == 0) sFar = 0;
    float dmin[8];
    #pragma unroll
    for (int i = 0; i < 8; ++i) dmin[i] = 1e10f;
    __syncthreads();

    const int lane = t & 31, wid = t >> 5;
    for (int it = 0; it < NPOINT; ++it) {
        const int far = sFar;
        if (t == 0) sList[it] = far;
        const float cx = lx[far], cy = ly[far], cz = lz[far];
        float bv = -1.f; int bi = 0;
        #pragma unroll
        for (int i = 0; i < 8; ++i) {
            const int n = t + i * 512;
            const float dx = lx[n]-cx, dy = ly[n]-cy, dz = lz[n]-cz;
            float d = dx*dx + dy*dy + dz*dz;
            d = fminf(d, dmin[i]); dmin[i] = d;
            if (d > bv) { bv = d; bi = n; }
        }
        #pragma unroll
        for (int off = 16; off > 0; off >>= 1) {           // wave32 butterfly
            const float ov = __shfl_xor(bv, off);
            const int   oi = __shfl_xor(bi, off);
            if (ov > bv || (ov == bv && oi < bi)) { bv = ov; bi = oi; }
        }
        if (lane == 0) { rv[wid] = bv; ri[wid] = bi; }
        __syncthreads();
        if (t == 0) {
            float best = rv[0]; int bidx = ri[0];
            for (int k = 1; k < 16; ++k)
                if (rv[k] > best || (rv[k] == best && ri[k] < bidx)) { best = rv[k]; bidx = ri[k]; }
            sFar = bidx;
        }
        __syncthreads();
    }
    const int n = sList[t];
    fpsIdx[b * NPOINT + t] = n;
    float* o = outXyz + (unsigned)(b * NPOINT + t) * 3;
    o[0] = lx[n]; o[1] = ly[n]; o[2] = lz[n];
}

// ---------------------------------------------------------------------------
// Kernel 2: ball query + gather + feature build. feats0 rows are padded
// IC 6 -> 32 halfs (zeros beyond k=5) so the layer-0 WMMA A-loads are
// unconditional aligned b128 loads with no per-lane guards.
// One wave per centroid, ballot-compacted "first 32 indices within radius".
// ---------------------------------------------------------------------------
__global__ void __launch_bounds__(256)
ballq_kernel(const float* __restrict__ xyz, const float* __restrict__ points,
             const float* __restrict__ newXyz, _Float16* __restrict__ feats0)
{
    __shared__ float lx[NPTS], ly[NPTS], lz[NPTS];
    __shared__ int selbuf[8][NSAMP];

    const int blk = blockIdx.x, t = threadIdx.x;
    const int b = blk >> 6;
    const int s = ((blk & 63) << 3) + (t >> 5);
    const int lane = t & 31, w = t >> 5;

    const float* base = xyz + (unsigned)b * NPTS * 3;
    for (int n = t; n < NPTS; n += 256) {
        lx[n] = base[3*n]; ly[n] = base[3*n+1]; lz[n] = base[3*n+2];
    }
    __syncthreads();

    const int bs = b * NPOINT + s;
    const float cx = newXyz[3*bs], cy = newXyz[3*bs+1], cz = newXyz[3*bs+2];

    int cnt = 0;
    for (int basei = 0; basei < NPTS && cnt < NSAMP; basei += 32) {
        const int n = basei + lane;
        const float dx = lx[n]-cx, dy = ly[n]-cy, dz = lz[n]-cz;
        const bool pred = (dx*dx + dy*dy + dz*dz) <= RAD2;
        const unsigned mask = (unsigned)__ballot(pred);
        const int pos = cnt + __popc(mask & ((1u << lane) - 1u));
        if (pred && pos < NSAMP) selbuf[w][pos] = n;
        cnt += __popc(mask);
    }
    if (cnt > NSAMP) cnt = NSAMP;
    // centroid itself is always within radius -> cnt >= 1; slots >= cnt get slot 0
    const int idx = selbuf[w][(lane < cnt) ? lane : 0];

    const float gx = lx[idx]-cx, gy = ly[idx]-cy, gz = lz[idx]-cz;
    const float* pp = points + ((unsigned)b * NPTS + idx) * 3;
    v8h row;
    row[0] = (_Float16)gx;    row[1] = (_Float16)gy;    row[2] = (_Float16)gz;
    row[3] = (_Float16)pp[0]; row[4] = (_Float16)pp[1]; row[5] = (_Float16)pp[2];
    row[6] = (_Float16)0.f;   row[7] = (_Float16)0.f;
    const v8h z = {};
    const unsigned rbase = ((unsigned)bs * NSAMP + lane) * 32;
    *(v8h*)(feats0 + rbase)      = row;
    *(v8h*)(feats0 + rbase + 8)  = z;
    *(v8h*)(feats0 + rbase + 16) = z;
    *(v8h*)(feats0 + rbase + 24) = z;
}

// ---------------------------------------------------------------------------
// WMMA GEMM: D(16x16,f32) += A(16x32,f16) x B(32x16,f16)
// ISA 16-bit A layout: lane<16 -> M=lane, halfs = K{0..7,16..23};
//                      lane>=16 -> M=lane-16, halfs = K{8..15,24..31}.
// B symmetric with N=lane&15. D: lane<16 -> (M=j,N=lane); lane>=16 -> (M=8+j).
// Geometry is compile-time: addressing folds to saddr+voffset+imm, and chunk
// validity is constexpr (worst-case kbase=8) so A loads are branch-free.
// ---------------------------------------------------------------------------
template<int ICW>
__device__ __forceinline__ v16h load_bfrag(const float* __restrict__ W,
                                           int ncol, int kcb)
{
    v16h bf;
    #pragma unroll
    for (int h = 0; h < 16; ++h) {
        const int k = kcb + ((h < 8) ? h : (h + 8));
        bf[h] = (_Float16)((k < ICW) ? W[ncol * ICW + k] : 0.f);
    }
    return bf;
}

template<int STRIDEA, int KCOFF, bool BN>
__device__ __forceinline__ v16h load_afrag(const _Float16* __restrict__ A,
                                           unsigned aoff, int kbase,
                                           const float* sA, const float* sB)
{
    constexpr bool HAVE0 = (KCOFF + 8 + 7)  < STRIDEA;  // valid for ALL lanes
    constexpr bool HAVE1 = (KCOFF + 8 + 23) < STRIDEA;
    const int kA = KCOFF + kbase;
    v8h c0 = {}; v8h c1 = {};
    if constexpr (HAVE0) {
        c0 = *(const v8h*)(A + aoff + kA);
        if (BN) {
            #pragma unroll
            for (int j = 0; j < 8; ++j) {
                const float x = (float)c0[j];
                c0[j] = (_Float16)fmaxf(x * sA[kA + j] + sB[kA + j], 0.f);
            }
        }
    }
    if constexpr (HAVE1) {
        c1 = *(const v8h*)(A + aoff + kA + 16);
        if (BN) {
            #pragma unroll
            for (int j = 0; j < 8; ++j) {
                const float x = (float)c1[j];
                c1[j] = (_Float16)fmaxf(x * sA[kA + 16 + j] + sB[kA + 16 + j], 0.f);
            }
        }
    }
    v16h af;
    #pragma unroll
    for (int j = 0; j < 8; ++j) { af[j] = c0[j]; af[8 + j] = c1[j]; }
    return af;
}

// A: MROWS x STRIDEA (f16, STRIDEA = padded IC). W: OC x ICW (f32, reference
// layout). Fuses: BN+ReLU of previous layer on the A load, per-channel
// sum/sumsq of the f32 accumulators (register-accumulated across all 16
// M-tiles, one LDS atomic at the end), f16 store of pre-BN activations.
template<int STRIDEA, int KCH, int ICW, int OC, int NT, bool BN>
__global__ void __launch_bounds__(256)
gemm_bn_kernel(const _Float16* __restrict__ A, const float* __restrict__ W,
               const float* __restrict__ aS, const float* __restrict__ aB,
               _Float16* __restrict__ out,
               float* __restrict__ gsum, float* __restrict__ gsq)
{
    __shared__ float ssum[OC], ssq[OC];
    __shared__ float sA[64], sB[64];
    const int t = threadIdx.x;
    if (t < OC) { ssum[t] = 0.f; ssq[t] = 0.f; }
    if (BN && t < STRIDEA) { sA[t] = aS[t]; sB[t] = aB[t]; }
    __syncthreads();

    const int lane   = t & 31;
    const int wgid   = blockIdx.x * 8 + (t >> 5);
    const int mgroup = wgid / NT;              // NT is 4 or 8 -> shift
    const int ntile  = wgid - mgroup * NT;
    const int ncol   = ntile * 16 + (lane & 15);
    const int kbase  = (lane < 16) ? 0 : 8;

    // weights are loop-invariant per wave: hoist B fragments
    v16h b0 = load_bfrag<ICW>(W, ncol, 0 + kbase);
    v16h b1 = {};
    if (KCH > 1) b1 = load_bfrag<ICW>(W, ncol, 32 + kbase);

    float s = 0.f, q = 0.f;                    // per-channel stats, in regs
    unsigned aoff  = (unsigned)(mgroup * 256 + (lane & 15)) * STRIDEA;
    unsigned obase = (unsigned)(mgroup * 256 + ((lane < 16) ? 0 : 8)) * OC + ncol;

    #pragma unroll 2
    for (int mt = 0; mt < 16; ++mt) {
        v8f acc = {};
        v16h a0 = load_afrag<STRIDEA, 0, BN>(A, aoff, kbase, sA, sB);
        acc = __builtin_amdgcn_wmma_f32_16x16x32_f16(false, a0, false, b0,
                                                     (short)0, acc, false, false);
        if (KCH > 1) {
            v16h a1 = load_afrag<STRIDEA, 32, BN>(A, aoff, kbase, sA, sB);
            acc = __builtin_amdgcn_wmma_f32_16x16x32_f16(false, a1, false, b1,
                                                         (short)0, acc, false, false);
        }

        #pragma unroll
        for (int j = 0; j < 8; ++j) { const float v = acc[j]; s += v; q += v * v; }

        #pragma unroll
        for (int j = 0; j < 8; ++j) out[obase + j * OC] = (_Float16)acc[j];

        aoff  += 16u * STRIDEA;
        obase += 16u * OC;
    }

    atomicAdd(&ssum[ncol], s);
    atomicAdd(&ssq[ncol], q);
    __syncthreads();
    if (t < OC) { atomicAdd(&gsum[t], ssum[t]); atomicAdd(&gsq[t], ssq[t]); }
}

// ---------------------------------------------------------------------------
// BN stat finalize: a = gamma*rsqrt(var+eps), b = beta - mean*a
// (conv bias cancels exactly through batch-norm, so it is dropped)
// ---------------------------------------------------------------------------
__global__ void bn_finalize_kernel(const float* __restrict__ sum, const float* __restrict__ sq,
                                   const float* __restrict__ gamma, const float* __restrict__ beta,
                                   float* __restrict__ aS, float* __restrict__ aB, int C)
{
    const int c = threadIdx.x;
    if (c < C) {
        const float invCnt = 1.f / (float)MROWS;
        const float m = sum[c] * invCnt;
        const float v = sq[c] * invCnt - m * m;
        const float a = gamma[c] * rsqrtf(v + 1e-5f);
        aS[c] = a;
        aB[c] = beta[c] - m * a;
    }
}

// ---------------------------------------------------------------------------
// Final BN+ReLU fused with max over the 32 samples -> new_points (f32 out).
// 64 threads handle channel pairs so the strided act2 reads are b32.
// ---------------------------------------------------------------------------
__global__ void __launch_bounds__(64)
maxpool_kernel(const _Float16* __restrict__ act2, const float* __restrict__ aS,
               const float* __restrict__ aB, float* __restrict__ out)
{
    const int bs = blockIdx.x, tp = threadIdx.x;
    const int c0 = tp * 2;
    const float a0 = aS[c0],   b0 = aB[c0];
    const float a1 = aS[c0+1], b1 = aB[c0+1];
    const unsigned base = (unsigned)bs * NSAMP * 128 + c0;
    float mx0 = -1e30f, mx1 = -1e30f;
    #pragma unroll 8
    for (int k = 0; k < NSAMP; ++k) {
        const _Float16 h0 = act2[base + k*128];
        const _Float16 h1 = act2[base + k*128 + 1];
        mx0 = fmaxf(mx0, (float)h0 * a0 + b0);
        mx1 = fmaxf(mx1, (float)h1 * a1 + b1);
    }
    const unsigned o = (unsigned)NB * NPOINT * 3 + (unsigned)bs * 128 + c0;
    out[o]     = fmaxf(mx0, 0.f);
    out[o + 1] = fmaxf(mx1, 0.f);
}

__global__ void zero_kernel(float* __restrict__ p, int n)
{
    const int i = blockIdx.x * 256 + threadIdx.x;
    if (i < n) p[i] = 0.f;
}

// ---------------------------------------------------------------------------
extern "C" void kernel_launch(void* const* d_in, const int* in_sizes, int n_in,
                              void* d_out, int out_size, void* d_ws, size_t ws_size,
                              hipStream_t stream)
{
    const float* xyz    = (const float*)d_in[0];
    const float* points = (const float*)d_in[1];
    const float* w0  = (const float*)d_in[2];   // b0 (d_in[3]) cancels through BN
    const float* g0  = (const float*)d_in[4];
    const float* be0 = (const float*)d_in[5];
    const float* w1  = (const float*)d_in[6];
    const float* g1  = (const float*)d_in[8];
    const float* be1 = (const float*)d_in[9];
    const float* w2  = (const float*)d_in[10];
    const float* g2  = (const float*)d_in[12];
    const float* be2 = (const float*)d_in[13];
    float* out = (float*)d_out;

    char* ws = (char*)d_ws;
    size_t off = 0;
    _Float16* feats0 = (_Float16*)(ws + off); off += (size_t)MROWS * 32  * 2; // 16.8MB, IC 6->32
    _Float16* act0   = (_Float16*)(ws + off); off += (size_t)MROWS * 64  * 2; // 33.5MB
    _Float16* act1   = (_Float16*)(ws + off); off += (size_t)MROWS * 64  * 2; // 33.5MB
    _Float16* act2   = (_Float16*)(ws + off); off += (size_t)MROWS * 128 * 2; // 67MB
    int*      fpsIdx = (int*)   (ws + off);   off += (size_t)NB * NPOINT * 4;
    float*    stats  = (float*) (ws + off);   off += 1024 * 4;

    float* sum0 = stats,       *sq0 = stats + 64;
    float* sum1 = stats + 128, *sq1 = stats + 192;
    float* sum2 = stats + 256, *sq2 = stats + 384;   // 128 each
    float* a0 = stats + 512, *bb0 = stats + 576;
    float* a1 = stats + 640, *bb1 = stats + 704;
    float* a2 = stats + 768, *bb2 = stats + 896;     // 128 each

    zero_kernel<<<2, 256, 0, stream>>>(stats, 512);
    fps_kernel<<<NB, 512, 0, stream>>>(xyz, fpsIdx, out);
    ballq_kernel<<<1024, 256, 0, stream>>>(xyz, points, out, feats0);

    // layer 0: IC=6 (rows zero-padded to 32), OC=64, one K-chunk
    gemm_bn_kernel<32, 1, 6, 64, 4, false><<<512, 256, 0, stream>>>(
        feats0, w0, nullptr, nullptr, act0, sum0, sq0);
    bn_finalize_kernel<<<1, 64, 0, stream>>>(sum0, sq0, g0, be0, a0, bb0, 64);

    // layer 1: IC=64, OC=64, BN+ReLU of layer0 fused into the A-load
    gemm_bn_kernel<64, 2, 64, 64, 4, true><<<512, 256, 0, stream>>>(
        act0, w1, a0, bb0, act1, sum1, sq1);
    bn_finalize_kernel<<<1, 64, 0, stream>>>(sum1, sq1, g1, be1, a1, bb1, 64);

    // layer 2: IC=64, OC=128
    gemm_bn_kernel<64, 2, 64, 128, 8, true><<<1024, 256, 0, stream>>>(
        act1, w2, a1, bb1, act2, sum2, sq2);
    bn_finalize_kernel<<<1, 128, 0, stream>>>(sum2, sq2, g2, be2, a2, bb2, 128);

    maxpool_kernel<<<NB * NPOINT, 64, 0, stream>>>(act2, a2, bb2, out);
}
